// nha_layer_21758304322128
// MI455X (gfx1250) — matrix operand order repaired and checked by reading, not compiled, not run on hardware
//
#include <hip/hip_runtime.h>

typedef __attribute__((ext_vector_type(16))) _Float16 v16h;
typedef __attribute__((ext_vector_type(8)))  _Float16 v8h;
typedef __attribute__((ext_vector_type(8)))  float    v8f;

#define WAVES   8
#define THREADS 256
#define BLOCKS  512
#define NWIN    16384
#define SCALE   0.17677669529663687f   // 1/sqrt(32)

__device__ __forceinline__ float sx(float v, int m) { return __shfl_xor(v, m, 32); }

__device__ __forceinline__ v8f wmma16(v16h a, v16h b, v8f c) {
  return __builtin_amdgcn_wmma_f32_16x16x32_f16(false, a, false, b, (short)0, c, false, false);
}

// A fragment (16x32, f16) from f32 global row-major [M][ld]
__device__ __forceinline__ v16h fragA_g(const float* base, int ld, int kb, int lane) {
  const int row = lane & 15, g = lane >> 4;
  const float* p = base + row * ld + kb + 8 * g;
  v16h r;
#pragma unroll
  for (int i = 0; i < 8; i++) { r[i] = (_Float16)p[i]; r[8 + i] = (_Float16)p[16 + i]; }
  return r;
}

// A fragment (16x32) from f16 LDS row-major [M][ld]
__device__ __forceinline__ v16h fragA_h(const _Float16* base, int ld, int kb, int lane) {
  const int row = lane & 15, g = lane >> 4;
  const _Float16* p = base + row * ld + kb + 8 * g;
  v8h lo = *(const v8h*)p;
  v8h hi = *(const v8h*)(p + 16);
  v16h r;
#pragma unroll
  for (int i = 0; i < 8; i++) { r[i] = lo[i]; r[8 + i] = hi[i]; }
  return r;
}

// B fragment (32x16) from f16 LDS holding B^T row-major [N][ldK]; baseT already at n-tile
__device__ __forceinline__ v16h fragB_h(const _Float16* baseT, int ld, int kb, int lane) {
  const int n = lane & 15, g = lane >> 4;
  const _Float16* p = baseT + n * ld + kb + 8 * g;
  v8h lo = *(const v8h*)p;
  v8h hi = *(const v8h*)(p + 16);
  v16h r;
#pragma unroll
  for (int i = 0; i < 8; i++) { r[i] = lo[i]; r[8 + i] = hi[i]; }
  return r;
}

// A fragment with only K=0..15 valid (att probabilities [16][16]), upper K zero
__device__ __forceinline__ v16h fragA_k16(const _Float16* base, int ld, int lane) {
  const int row = lane & 15, g = lane >> 4;
  const _Float16* p = base + row * ld + 8 * g;
  v8h lo = *(const v8h*)p;
  v16h r;
#pragma unroll
  for (int i = 0; i < 8; i++) { r[i] = lo[i]; r[8 + i] = (_Float16)0.0f; }
  return r;
}

// B fragment with only K=0..15 valid (from v^T [N][16]); elements 8-15 carry K=16..31 -> zero
__device__ __forceinline__ v16h fragB_k16(const _Float16* base, int ld, int lane) {
  const int n = lane & 15, g = lane >> 4;
  v16h r;
#pragma unroll
  for (int i = 0; i < 16; i++) r[i] = (_Float16)0.0f;
  {
    const _Float16* p = base + n * ld + 8 * g;
    v8h lo = *(const v8h*)p;
#pragma unroll
    for (int i = 0; i < 8; i++) { r[i] = lo[i]; }
  }
  return r;
}

// LayerNorm over 8 C-tiles (full 128-feature rows), stats via 16-lane xor reductions
__device__ __forceinline__ void ln8(v8f* a, const float* gamma, const float* beta, int lane) {
  const int col16 = lane & 15;
#pragma unroll
  for (int v = 0; v < 8; v++) {
    float s = 0.f, s2 = 0.f;
#pragma unroll
    for (int nt = 0; nt < 8; nt++) { float t = a[nt][v]; s += t; s2 += t * t; }
#pragma unroll
    for (int m = 1; m < 16; m <<= 1) { s += sx(s, m); s2 += sx(s2, m); }
    float mean = s * 0.0078125f;
    float var  = s2 * 0.0078125f - mean * mean;
    float inv  = rsqrtf(var + 1e-5f);
#pragma unroll
    for (int nt = 0; nt < 8; nt++) {
      int c = nt * 16 + col16;
      a[nt][v] = (a[nt][v] - mean) * inv * gamma[c] + beta[c];
    }
  }
}

// ---------------- Kernel 1: input proj + LN + biased MHA + residual + LN -> y (d_out) --------
__global__ __launch_bounds__(THREADS)
void nha_attn_kernel(const float* __restrict__ x, const float* __restrict__ pos1,
                     const float* __restrict__ pos2, const float* __restrict__ W_in,
                     const float* __restrict__ g_in, const float* __restrict__ b_in,
                     const float* __restrict__ W_pe, const float* __restrict__ W_bias,
                     const float* __restrict__ Wq, const float* __restrict__ bq,
                     const float* __restrict__ Wk, const float* __restrict__ bk,
                     const float* __restrict__ Wv, const float* __restrict__ bv,
                     const float* __restrict__ Wo, const float* __restrict__ bo,
                     const float* __restrict__ g1, const float* __restrict__ b1,
                     float* __restrict__ out) {
  __shared__ _Float16 sWin[128 * 64];                    // W_in^T [out][in]
  __shared__ _Float16 sWq[128 * 128], sWk[128 * 128];    // W^T [out][in]
  __shared__ _Float16 sWv[128 * 128], sWo[128 * 128];
  __shared__ float    sP[1216];                          // biases, LN params, W_pe, W_bias
  __shared__ _Float16 sScr[WAVES][3 * 16 * 128];         // per-wave A / Q / K scratch

  const int tid = threadIdx.x;
  for (int e = tid; e < 128 * 64; e += THREADS)
    sWin[e] = (_Float16)W_in[(e & 63) * 128 + (e >> 6)];
  for (int e = tid; e < 128 * 128; e += THREADS) {
    int o = e >> 7, i = e & 127;
    sWq[e] = (_Float16)Wq[i * 128 + o];
    sWk[e] = (_Float16)Wk[i * 128 + o];
    sWv[e] = (_Float16)Wv[i * 128 + o];
    sWo[e] = (_Float16)Wo[i * 128 + o];
  }
  if (tid < 128) {
    sP[tid] = bq[tid];        sP[128 + tid] = bk[tid];
    sP[256 + tid] = bv[tid];  sP[384 + tid] = bo[tid];
    sP[512 + tid] = g_in[tid]; sP[640 + tid] = b_in[tid];
    sP[768 + tid] = g1[tid];  sP[896 + tid] = b1[tid];
    sP[1088 + tid] = W_bias[tid];
  }
  if (tid < 64) sP[1024 + tid] = W_pe[tid];
  __syncthreads();

  const int wave = tid >> 5, lane = tid & 31;
  const int col16 = lane & 15, rb = (lane >> 4) * 8;
  _Float16* bufA = sScr[wave];
  _Float16* bufQ = bufA + 2048;
  _Float16* bufK = bufA + 4096;

  const int wstart = blockIdx.x * WAVES + wave;
  const int wstep  = gridDim.x * WAVES;
  for (int w = wstart; w < NWIN; w += wstep) {
    if (w + wstep < NWIN) {
      __builtin_prefetch(x + (size_t)(w + wstep) * 1024, 0, 0);
      __builtin_prefetch(pos1 + (size_t)(w + wstep) * 256, 0, 0);
      __builtin_prefetch(pos2 + (size_t)(w + wstep) * 256, 0, 0);
    }
    const float* xw = x + (size_t)w * 1024;

    // ---- h = LN(x @ W_in) -> bufA (t, f16 row-major [16][128]) ----
    v16h xf0 = fragA_g(xw, 64, 0, lane);
    v16h xf1 = fragA_g(xw, 64, 32, lane);
    v8f acc[8];
#pragma unroll
    for (int nt = 0; nt < 8; nt++) {
      v8f c = {};
      c = wmma16(xf0, fragB_h(sWin + nt * 16 * 64, 64, 0, lane), c);
      c = wmma16(xf1, fragB_h(sWin + nt * 16 * 64, 64, 32, lane), c);
      acc[nt] = c;
    }
    ln8(acc, sP + 512, sP + 640, lane);
#pragma unroll
    for (int nt = 0; nt < 8; nt++)
#pragma unroll
      for (int v = 0; v < 8; v++)
        bufA[(v + rb) * 128 + nt * 16 + col16] = (_Float16)acc[nt][v];

    // ---- q, k projections -> bufQ / bufK ----
    v16h at[4];
#pragma unroll
    for (int kt = 0; kt < 4; kt++) at[kt] = fragA_h(bufA, 128, kt * 32, lane);
#pragma unroll
    for (int nt = 0; nt < 8; nt++) {
      v8f c = {};
#pragma unroll
      for (int kt = 0; kt < 4; kt++)
        c = wmma16(at[kt], fragB_h(sWq + nt * 16 * 128, 128, kt * 32, lane), c);
      float bb = sP[nt * 16 + col16];
#pragma unroll
      for (int v = 0; v < 8; v++)
        bufQ[(v + rb) * 128 + nt * 16 + col16] = (_Float16)(c[v] + bb);
    }
#pragma unroll
    for (int nt = 0; nt < 8; nt++) {
      v8f c = {};
#pragma unroll
      for (int kt = 0; kt < 4; kt++)
        c = wmma16(at[kt], fragB_h(sWk + nt * 16 * 128, 128, kt * 32, lane), c);
      float bb = sP[128 + nt * 16 + col16];
#pragma unroll
      for (int v = 0; v < 8; v++)
        bufK[(v + rb) * 128 + nt * 16 + col16] = (_Float16)(c[v] + bb);
    }

    // ---- relative-position bias: each lane computes its own score-layout cells ----
    float bs[4][8];
#pragma unroll
    for (int v = 0; v < 8; v++) {
      const int qrow = v + rb;
      float p1 = pos1[(size_t)w * 256 + qrow * 16 + col16];
      float p2 = pos2[(size_t)w * 256 + qrow * 16 + col16];
      float a0 = 0.f, a1 = 0.f, a2 = 0.f, a3 = 0.f;
      for (int j = 0; j < 32; j++) {
        float t = p1 * sP[1024 + j] + p2 * sP[1056 + j];
        float e = t * (1.0f / (1.0f + __expf(-t)));      // SiLU
        a0 += e * sP[1088 + 4 * j + 0];
        a1 += e * sP[1088 + 4 * j + 1];
        a2 += e * sP[1088 + 4 * j + 2];
        a3 += e * sP[1088 + 4 * j + 3];
      }
      bs[0][v] = a0; bs[1][v] = a1; bs[2][v] = a2; bs[3][v] = a3;
    }

    // ---- scores per head (q @ k^T), then softmax; att stored into bufQ ----
    v8f sacc[4];
#pragma unroll
    for (int h = 0; h < 4; h++) {
      v8f c = {};
      sacc[h] = wmma16(fragA_h(bufQ, 128, h * 32, lane),
                       fragB_h(bufK, 128, h * 32, lane), c);
    }
#pragma unroll
    for (int h = 0; h < 4; h++) {
      float pr[8];
#pragma unroll
      for (int v = 0; v < 8; v++) pr[v] = sacc[h][v] * SCALE + bs[h][v];
#pragma unroll
      for (int v = 0; v < 8; v++) {
        float m = pr[v];
        m = fmaxf(m, sx(m, 1)); m = fmaxf(m, sx(m, 2));
        m = fmaxf(m, sx(m, 4)); m = fmaxf(m, sx(m, 8));
        pr[v] = __expf(pr[v] - m);
      }
#pragma unroll
      for (int v = 0; v < 8; v++) {
        float s = pr[v];
        s += sx(s, 1); s += sx(s, 2); s += sx(s, 4); s += sx(s, 8);
        pr[v] *= (1.0f / s);
      }
#pragma unroll
      for (int v = 0; v < 8; v++)
        bufQ[h * 256 + (v + rb) * 16 + col16] = (_Float16)pr[v];
    }

    // ---- v projection -> v^T into bufK ([feature][token]) ----
#pragma unroll
    for (int nt = 0; nt < 8; nt++) {
      v8f c = {};
#pragma unroll
      for (int kt = 0; kt < 4; kt++)
        c = wmma16(at[kt], fragB_h(sWv + nt * 16 * 128, 128, kt * 32, lane), c);
      float bb = sP[256 + nt * 16 + col16];
#pragma unroll
      for (int v = 0; v < 8; v++)
        bufK[(nt * 16 + col16) * 16 + (v + rb)] = (_Float16)(c[v] + bb);
    }

    // ---- o = att @ v (K=16 padded to 32) ----
    v8f oacc[8];
#pragma unroll
    for (int h = 0; h < 4; h++) {
      v16h af = fragA_k16(bufQ + h * 256, 16, lane);
#pragma unroll
      for (int dt = 0; dt < 2; dt++) {
        v8f c = {};
        oacc[h * 2 + dt] = wmma16(af, fragB_k16(bufK + (h * 32 + dt * 16) * 16, 16, lane), c);
      }
    }
#pragma unroll
    for (int nt = 0; nt < 8; nt++)
#pragma unroll
      for (int v = 0; v < 8; v++)
        bufQ[(v + rb) * 128 + nt * 16 + col16] = (_Float16)oacc[nt][v];

    // ---- att_out = o @ Wo + bo ; y = LN(t + att_out) -> global (d_out) ----
    v16h ao[4];
#pragma unroll
    for (int kt = 0; kt < 4; kt++) ao[kt] = fragA_h(bufQ, 128, kt * 32, lane);
    v8f yacc[8];
#pragma unroll
    for (int nt = 0; nt < 8; nt++) {
      v8f c = {};
#pragma unroll
      for (int kt = 0; kt < 4; kt++)
        c = wmma16(ao[kt], fragB_h(sWo + nt * 16 * 128, 128, kt * 32, lane), c);
      float bb = sP[384 + nt * 16 + col16];
#pragma unroll
      for (int v = 0; v < 8; v++)
        c[v] += bb + (float)bufA[(v + rb) * 128 + nt * 16 + col16];
      yacc[nt] = c;
    }
    ln8(yacc, sP + 768, sP + 896, lane);
    float* outw = out + (size_t)w * 2048;
#pragma unroll
    for (int nt = 0; nt < 8; nt++)
#pragma unroll
      for (int v = 0; v < 8; v++)
        outw[(v + rb) * 128 + nt * 16 + col16] = yacc[nt][v];
  }
}

// ---------------- Kernel 2: y -> LN(y + silu(y@W1)@W2) in-place on d_out --------------------
__global__ __launch_bounds__(THREADS)
void nha_ffn_kernel(const float* __restrict__ W1, const float* __restrict__ W2,
                    const float* __restrict__ g2, const float* __restrict__ b2,
                    float* __restrict__ out) {
  __shared__ _Float16 sW1t[256 * 128];          // W1^T [FF][MD]
  __shared__ _Float16 sW2t[128 * 256];          // W2^T [MD][FF]
  __shared__ float    sQ[256];
  __shared__ _Float16 sU[WAVES][16 * 256];      // per-wave silu(y@W1) staging

  const int tid = threadIdx.x;
  for (int e = tid; e < 256 * 128; e += THREADS) {
    int o = e >> 7, i = e & 127;
    sW1t[e] = (_Float16)W1[i * 256 + o];
  }
  for (int e = tid; e < 128 * 256; e += THREADS) {
    int o = e >> 8, i = e & 255;
    sW2t[e] = (_Float16)W2[i * 128 + o];
  }
  if (tid < 128) { sQ[tid] = g2[tid]; sQ[128 + tid] = b2[tid]; }
  __syncthreads();

  const int wave = tid >> 5, lane = tid & 31;
  const int col16 = lane & 15, rb = (lane >> 4) * 8;
  _Float16* bufU = sU[wave];

  const int wstart = blockIdx.x * WAVES + wave;
  const int wstep  = gridDim.x * WAVES;
  for (int w = wstart; w < NWIN; w += wstep) {
    float* yw = out + (size_t)w * 2048;
    if (w + wstep < NWIN) __builtin_prefetch(out + (size_t)(w + wstep) * 2048, 0, 0);

    v16h ay[4];
#pragma unroll
    for (int kt = 0; kt < 4; kt++) ay[kt] = fragA_g(yw, 128, kt * 32, lane);

    // u = silu(y @ W1) -> bufU (f16 [16][256])
#pragma unroll
    for (int nt = 0; nt < 16; nt++) {
      v8f c = {};
#pragma unroll
      for (int kt = 0; kt < 4; kt++)
        c = wmma16(ay[kt], fragB_h(sW1t + nt * 16 * 128, 128, kt * 32, lane), c);
#pragma unroll
      for (int v = 0; v < 8; v++) {
        float t = c[v];
        bufU[(v + rb) * 256 + nt * 16 + col16] = (_Float16)(t * (1.0f / (1.0f + __expf(-t))));
      }
    }

    // z = u @ W2 + y ; out = LN(z)
    v16h au[8];
#pragma unroll
    for (int kt = 0; kt < 8; kt++) au[kt] = fragA_h(bufU, 256, kt * 32, lane);
    v8f z[8];
#pragma unroll
    for (int nt = 0; nt < 8; nt++) {
      v8f c = {};
#pragma unroll
      for (int kt = 0; kt < 8; kt++)
        c = wmma16(au[kt], fragB_h(sW2t + nt * 16 * 256, 256, kt * 32, lane), c);
#pragma unroll
      for (int v = 0; v < 8; v++)
        c[v] += yw[(v + rb) * 128 + nt * 16 + col16];
      z[nt] = c;
    }
    ln8(z, sQ, sQ + 128, lane);
#pragma unroll
    for (int nt = 0; nt < 8; nt++)
#pragma unroll
      for (int v = 0; v < 8; v++)
        yw[(v + rb) * 128 + nt * 16 + col16] = z[nt][v];
  }
}

extern "C" void kernel_launch(void* const* d_in, const int* in_sizes, int n_in,
                              void* d_out, int out_size, void* d_ws, size_t ws_size,
                              hipStream_t stream) {
  (void)in_sizes; (void)n_in; (void)out_size; (void)d_ws; (void)ws_size;
  const float* x      = (const float*)d_in[0];
  const float* pos1   = (const float*)d_in[1];
  const float* pos2   = (const float*)d_in[2];
  const float* W_in   = (const float*)d_in[3];
  const float* g_in   = (const float*)d_in[4];
  const float* b_in   = (const float*)d_in[5];
  const float* W_pe   = (const float*)d_in[6];
  const float* W_bias = (const float*)d_in[7];
  const float* Wq     = (const float*)d_in[8];
  const float* bq     = (const float*)d_in[9];
  const float* Wk     = (const float*)d_in[10];
  const float* bk     = (const float*)d_in[11];
  const float* Wv     = (const float*)d_in[12];
  const float* bv     = (const float*)d_in[13];
  const float* Wo     = (const float*)d_in[14];
  const float* bo     = (const float*)d_in[15];
  const float* W1     = (const float*)d_in[16];
  const float* W2     = (const float*)d_in[17];
  const float* g1     = (const float*)d_in[18];
  const float* b1     = (const float*)d_in[19];
  const float* g2     = (const float*)d_in[20];
  const float* b2     = (const float*)d_in[21];
  float* out = (float*)d_out;

  nha_attn_kernel<<<BLOCKS, THREADS, 0, stream>>>(
      x, pos1, pos2, W_in, g_in, b_in, W_pe, W_bias,
      Wq, bq, Wk, bk, Wv, bv, Wo, bo, g1, b1, out);
  nha_ffn_kernel<<<BLOCKS, THREADS, 0, stream>>>(W1, W2, g2, b2, out);
}
